// LinearAttention_51316269252717
// MI455X (gfx1250) — matrix operand order repaired and checked by reading, not compiled, not run on hardware
//
#include <hip/hip_runtime.h>

typedef __bf16 bf16_t;
typedef __attribute__((ext_vector_type(16))) __bf16 v16bf;
typedef __attribute__((ext_vector_type(8)))  __bf16 v8bf;
typedef __attribute__((ext_vector_type(8)))  float  v8f;

namespace {
constexpr int kDim   = 1024;
constexpr int kHeads = 16;
constexpr int kHd    = 64;
constexpr int kB     = 4;
constexpr int kN     = 4096;   // sequence length = H*W
constexpr int kHW    = 64;     // H = W = 64
constexpr int kM     = kB * kN;       // 16384 total rows
constexpr int kBH    = kB * kHeads;   // 64 (b,h) pairs
constexpr float kEps = 1e-6f;
}

__device__ __forceinline__ bf16_t f2bf(float f) {
  unsigned u = __float_as_uint(f);
  u += 0x7fffu + ((u >> 16) & 1u);           // round-to-nearest-even
  return __builtin_bit_cast(bf16_t, (unsigned short)(u >> 16));
}
__device__ __forceinline__ float bf2f(bf16_t b) {
  return __uint_as_float(((unsigned)__builtin_bit_cast(unsigned short, b)) << 16);
}

// ---------------------------------------------------------------------------
// WMMA fragment loaders (wave32, V_WMMA_F32_16X16X32_BF16 layouts).
// A (16x32, MxK): lane L -> row row0+(L&15); K = {b8..b8+7} U {16+b8..16+b8+7},
// b8 = 8*(L>>4). Two contiguous 16B loads from a row-major [M][K] array.
// ---------------------------------------------------------------------------
__device__ __forceinline__ v16bf load_frag_a(const bf16_t* __restrict__ p, int ld,
                                             int row0, int k0) {
  int lane = threadIdx.x & 31;
  const bf16_t* base =
      p + (size_t)(row0 + (lane & 15)) * ld + k0 + ((lane >> 4) << 3);
  v8bf lo = *(const v8bf*)base;
  v8bf hi = *(const v8bf*)(base + 16);
  return __builtin_shufflevector(lo, hi, 0, 1, 2, 3, 4, 5, 6, 7,
                                 8, 9, 10, 11, 12, 13, 14, 15);
}

// B (32x16, KxN) from an N-major array BT[ncol][K]: lane L -> col col0+(L&15);
// K = {b8..b8+7} U {16+b8..16+b8+7}, b8 = 8*(L>>4) (same order as A).
__device__ __forceinline__ v16bf load_frag_b(const bf16_t* __restrict__ p, int ld,
                                             int col0, int k0) {
  int lane = threadIdx.x & 31;
  const bf16_t* base =
      p + (size_t)(col0 + (lane & 15)) * ld + k0 + ((lane >> 4) << 3);
  v8bf lo = *(const v8bf*)base;
  v8bf hi = *(const v8bf*)(base + 16);
  return __builtin_shufflevector(lo, hi, 0, 1, 2, 3, 4, 5, 6, 7,
                                 8, 9, 10, 11, 12, 13, 14, 15);
}

#define WMMA_BF16(A, B, C) \
  __builtin_amdgcn_wmma_f32_16x16x32_bf16(false, (A), false, (B), (short)0, (C), \
                                          false, false)

// ---------------------------------------------------------------------------
// Elementwise fp32 -> bf16 convert (x).
// ---------------------------------------------------------------------------
__global__ void cvt_kernel(const float* __restrict__ src, bf16_t* __restrict__ dst,
                           int nElem) {
  int i = blockIdx.x * 256 + threadIdx.x;
  if (i < nElem) dst[i] = f2bf(src[i]);
}

// Transpose + convert: W[K][Ncol] fp32 -> WT[Ncol][K] bf16 (N-major for B frags).
__global__ void transpose_cvt_kernel(const float* __restrict__ W,
                                     bf16_t* __restrict__ WT, int K, int Ncol) {
  int idx = blockIdx.x * 256 + threadIdx.x;
  if (idx < K * Ncol) {
    int k = idx / Ncol;
    int n = idx - k * Ncol;
    WT[(size_t)n * K + k] = f2bf(W[idx]);
  }
}

// ---------------------------------------------------------------------------
// Fused qkv GEMM: [16384 x 1024] @ [1024 x 3072] with relu+eps on q,k.
// q -> qb [bh][N][d] row-major; k,v -> kT/vT [bh][d][N] (transposed planes).
// One 64x64 tile per wave; 4 waves / block.
// ---------------------------------------------------------------------------
__global__ __launch_bounds__(128) void gemm_qkv_kernel(
    const bf16_t* __restrict__ xb, const bf16_t* __restrict__ wqT,
    const bf16_t* __restrict__ wkvT, bf16_t* __restrict__ qb,
    bf16_t* __restrict__ kT, bf16_t* __restrict__ vT) {
  int wave = threadIdx.x >> 5;
  int lane = threadIdx.x & 31;
  int t = blockIdx.x * 4 + wave;     // 12288 tiles: 256 row x 48 col
  int m0 = (t & 255) << 6;
  int c0 = (t >> 8) << 6;
  const bf16_t* wt;
  int cw0;
  if (c0 < kDim) { wt = wqT; cw0 = c0; } else { wt = wkvT; cw0 = c0 - kDim; }

  v8f acc[4][4] = {};
  for (int k0 = 0; k0 < kDim; k0 += 32) {
    v16bf a[4], b[4];
#pragma unroll
    for (int i = 0; i < 4; ++i) a[i] = load_frag_a(xb, kDim, m0 + i * 16, k0);
#pragma unroll
    for (int j = 0; j < 4; ++j) b[j] = load_frag_b(wt, kDim, cw0 + j * 16, k0);
    if (k0 + 64 < kDim) {  // L2 prefetch of next-next K slab (global_prefetch_b8)
      __builtin_prefetch(xb + (size_t)(m0 + (lane & 15)) * kDim + k0 + 64, 0, 3);
      __builtin_prefetch(wt + (size_t)(cw0 + (lane & 15)) * kDim + k0 + 64, 0, 3);
    }
#pragma unroll
    for (int i = 0; i < 4; ++i)
#pragma unroll
      for (int j = 0; j < 4; ++j)
        acc[i][j] = WMMA_BF16(a[i], b[j], acc[i][j]);
  }

#pragma unroll
  for (int i = 0; i < 4; ++i) {
#pragma unroll
    for (int j = 0; j < 4; ++j) {
      int c = c0 + j * 16 + (lane & 15);
#pragma unroll
      for (int r = 0; r < 8; ++r) {
        int m = m0 + i * 16 + ((lane >> 4) << 3) + r;
        float v = acc[i][j][r];
        int b_ = m >> 12;          // batch
        int n = m & (kN - 1);      // position
        if (c < kDim) {            // q
          v = fmaxf(v, 0.f) + kEps;
          int hh = c >> 6, dd = c & 63;
          qb[(((size_t)(b_ * kHeads + hh) * kN + n) << 6) + dd] = f2bf(v);
        } else if (c < 2 * kDim) { // k (store transposed [bh][d][N])
          int cc = c - kDim;
          v = fmaxf(v, 0.f) + kEps;
          kT[((size_t)(b_ * kHeads + (cc >> 6)) * kHd + (cc & 63)) * kN + n] =
              f2bf(v);
        } else {                   // v (store transposed [bh][d][N])
          int cc = c - 2 * kDim;
          vT[((size_t)(b_ * kHeads + (cc >> 6)) * kHd + (cc & 63)) * kN + n] =
              f2bf(v);
        }
      }
    }
  }
}

// ---------------------------------------------------------------------------
// k_mean[bh][d] = mean_n k[bh][n][d]  (kT planes are contiguous in n).
// ---------------------------------------------------------------------------
__global__ void kmean_kernel(const bf16_t* __restrict__ kT,
                             float* __restrict__ kmean) {
  __shared__ float red[256];
  int bh = blockIdx.x;
  int dd = threadIdx.x & 63;
  int seg = threadIdx.x >> 6;
  const bf16_t* row = kT + ((size_t)bh * kHd + dd) * kN + seg * (kN / 4);
  float s = 0.f;
  for (int n = 0; n < kN / 4; ++n) s += bf2f(row[n]);
  red[threadIdx.x] = s;
  __syncthreads();
  if (seg == 0) {
    float tsum = red[dd] + red[dd + 64] + red[dd + 128] + red[dd + 192];
    kmean[bh * kHd + dd] = tsum * (1.f / kN);
  }
}

// z[bh][n] = 1 / (q[bh][n][:] . kmean[bh][:] + eps)
__global__ void z_kernel(const bf16_t* __restrict__ qb,
                         const float* __restrict__ kmean,
                         float* __restrict__ zb) {
  int bh = blockIdx.x >> 4;
  int n = ((blockIdx.x & 15) << 8) + threadIdx.x;
  const bf16_t* qrow = qb + ((size_t)bh * kN + n) * kHd;
  const float* km = kmean + bh * kHd;
  float s = 0.f;
#pragma unroll 8
  for (int d = 0; d < kHd; ++d) s += bf2f(qrow[d]) * km[d];
  zb[bh * kN + n] = 1.f / (s + kEps);
}

// ---------------------------------------------------------------------------
// kv[bh] = (1/N) k^T v  (64x64, K=4096). 4 waves split K; reduce through LDS.
// Output stored transposed kvT[bh][e][d] so it is a ready WMMA B operand.
// ---------------------------------------------------------------------------
__global__ __launch_bounds__(128) void kv_kernel(const bf16_t* __restrict__ kT,
                                                 const bf16_t* __restrict__ vT,
                                                 bf16_t* __restrict__ kvT) {
  __shared__ float red[kHd * kHd];
  int bh = blockIdx.x;
  int wave = threadIdx.x >> 5;
  int lane = threadIdx.x & 31;
  const bf16_t* kA = kT + (size_t)bh * kHd * kN;  // A[d][n]
  const bf16_t* vB = vT + (size_t)bh * kHd * kN;  // BT[e][n]
  for (int i = threadIdx.x; i < kHd * kHd; i += 128) red[i] = 0.f;
  __syncthreads();

  v8f acc[4][4] = {};
  int nbase = wave * (kN / 4);
  for (int k0 = 0; k0 < kN / 4; k0 += 32) {
    v16bf a[4], b[4];
#pragma unroll
    for (int i = 0; i < 4; ++i) a[i] = load_frag_a(kA, kN, i * 16, nbase + k0);
#pragma unroll
    for (int j = 0; j < 4; ++j) b[j] = load_frag_b(vB, kN, j * 16, nbase + k0);
#pragma unroll
    for (int i = 0; i < 4; ++i)
#pragma unroll
      for (int j = 0; j < 4; ++j)
        acc[i][j] = WMMA_BF16(a[i], b[j], acc[i][j]);
  }

#pragma unroll
  for (int i = 0; i < 4; ++i)
#pragma unroll
    for (int j = 0; j < 4; ++j) {
      int e = j * 16 + (lane & 15);
#pragma unroll
      for (int r = 0; r < 8; ++r) {
        int dd = i * 16 + ((lane >> 4) << 3) + r;
        atomicAdd(&red[e * kHd + dd], acc[i][j][r]);  // ds_add_f32
      }
    }
  __syncthreads();
  for (int i = threadIdx.x; i < kHd * kHd; i += 128)
    kvT[(size_t)bh * kHd * kHd + i] = f2bf(red[i] * (1.f / kN));
}

// ---------------------------------------------------------------------------
// Depthwise 5x5 conv over vT channel planes: cb[bh][d][n] = conv(v)[...] + bias
// ---------------------------------------------------------------------------
__global__ void dwconv_kernel(const bf16_t* __restrict__ vT,
                              const float* __restrict__ w,
                              const float* __restrict__ bias,
                              bf16_t* __restrict__ cb) {
  int idx = blockIdx.x * 256 + threadIdx.x;  // (bh*64+dd)*4096 + n
  int n = idx & (kN - 1);
  int dd = (idx >> 12) & 63;
  int x = n & 63, y = n >> 6;
  const bf16_t* plane = vT + ((size_t)(idx >> 12)) * kN;
  const float* wk = w + dd * 25;
  float s = bias[dd];
#pragma unroll
  for (int ky = 0; ky < 5; ++ky) {
    int yy = y + ky - 2;
    if (yy < 0 || yy >= kHW) continue;
#pragma unroll
    for (int kx = 0; kx < 5; ++kx) {
      int xx = x + kx - 2;
      if (xx < 0 || xx >= kHW) continue;
      s += bf2f(plane[yy * kHW + xx]) * wk[ky * 5 + kx];
    }
  }
  cb[idx] = f2bf(s);
}

// ---------------------------------------------------------------------------
// att = (q @ kv) * z + conv, written as y[B][N][C] bf16. K = 64 (2 WMMA steps).
// ---------------------------------------------------------------------------
__global__ __launch_bounds__(128) void att_kernel(
    const bf16_t* __restrict__ qb, const bf16_t* __restrict__ kvT,
    const float* __restrict__ zb, const bf16_t* __restrict__ cb,
    bf16_t* __restrict__ y) {
  int bh = blockIdx.x;
  int wave = threadIdx.x >> 5;
  int lane = threadIdx.x & 31;
  int m0 = (blockIdx.y * 4 + wave) << 6;
  const bf16_t* qA = qb + (size_t)bh * kN * kHd;     // [N][64]
  const bf16_t* kvB = kvT + (size_t)bh * kHd * kHd;  // [e][64]

  v8f acc[4][4] = {};
#pragma unroll
  for (int k0 = 0; k0 < kHd; k0 += 32) {
    v16bf a[4], b[4];
#pragma unroll
    for (int i = 0; i < 4; ++i) a[i] = load_frag_a(qA, kHd, m0 + i * 16, k0);
#pragma unroll
    for (int j = 0; j < 4; ++j) b[j] = load_frag_b(kvB, kHd, j * 16, k0);
#pragma unroll
    for (int i = 0; i < 4; ++i)
#pragma unroll
      for (int j = 0; j < 4; ++j)
        acc[i][j] = WMMA_BF16(a[i], b[j], acc[i][j]);
  }

  int b_ = bh >> 4, hh = bh & 15;
#pragma unroll
  for (int i = 0; i < 4; ++i)
#pragma unroll
    for (int j = 0; j < 4; ++j) {
      int e = j * 16 + (lane & 15);
#pragma unroll
      for (int r = 0; r < 8; ++r) {
        int n = m0 + i * 16 + ((lane >> 4) << 3) + r;
        float v = acc[i][j][r] * zb[bh * kN + n] +
                  bf2f(cb[((size_t)bh * kHd + e) * kN + n]);
        y[(((size_t)b_ * kN + n) << 10) + (hh << 6) + e] = f2bf(v);
      }
    }
}

// ---------------------------------------------------------------------------
// out = y @ Wproj + bproj  (fp32 output).
// ---------------------------------------------------------------------------
__global__ __launch_bounds__(128) void gemm_out_kernel(
    const bf16_t* __restrict__ yb, const bf16_t* __restrict__ wpT,
    const float* __restrict__ bias, float* __restrict__ out) {
  int wave = threadIdx.x >> 5;
  int lane = threadIdx.x & 31;
  int t = blockIdx.x * 4 + wave;  // 4096 tiles: 256 row x 16 col
  int m0 = (t & 255) << 6;
  int c0 = (t >> 8) << 6;

  v8f acc[4][4] = {};
  for (int k0 = 0; k0 < kDim; k0 += 32) {
    v16bf a[4], b[4];
#pragma unroll
    for (int i = 0; i < 4; ++i) a[i] = load_frag_a(yb, kDim, m0 + i * 16, k0);
#pragma unroll
    for (int j = 0; j < 4; ++j) b[j] = load_frag_b(wpT, kDim, c0 + j * 16, k0);
    if (k0 + 64 < kDim) {
      __builtin_prefetch(yb + (size_t)(m0 + (lane & 15)) * kDim + k0 + 64, 0, 3);
      __builtin_prefetch(wpT + (size_t)(c0 + (lane & 15)) * kDim + k0 + 64, 0, 3);
    }
#pragma unroll
    for (int i = 0; i < 4; ++i)
#pragma unroll
      for (int j = 0; j < 4; ++j)
        acc[i][j] = WMMA_BF16(a[i], b[j], acc[i][j]);
  }

#pragma unroll
  for (int i = 0; i < 4; ++i)
#pragma unroll
    for (int j = 0; j < 4; ++j) {
      int c = c0 + j * 16 + (lane & 15);
      float bv = bias[c];
#pragma unroll
      for (int r = 0; r < 8; ++r) {
        int m = m0 + i * 16 + ((lane >> 4) << 3) + r;
        out[(size_t)m * kDim + c] = acc[i][j][r] + bv;
      }
    }
}

// ---------------------------------------------------------------------------
extern "C" void kernel_launch(void* const* d_in, const int* in_sizes, int n_in,
                              void* d_out, int out_size, void* d_ws,
                              size_t ws_size, hipStream_t stream) {
  const float* x = (const float*)d_in[0];
  const float* Wq = (const float*)d_in[1];
  const float* Wkv = (const float*)d_in[2];
  const float* Wproj = (const float*)d_in[3];
  const float* bproj = (const float*)d_in[4];
  const float* dwc_w = (const float*)d_in[5];
  const float* dwc_b = (const float*)d_in[6];
  float* out = (float*)d_out;

  char* p = (char*)d_ws;
  auto alloc = [&](size_t bytes) -> char* {
    char* r = p;
    p += (bytes + 255) & ~(size_t)255;
    return r;
  };
  bf16_t* xb = (bf16_t*)alloc((size_t)kM * kDim * 2);          // x bf16
  bf16_t* wqT = (bf16_t*)alloc((size_t)kDim * kDim * 2);       // Wq^T
  bf16_t* wkvT = (bf16_t*)alloc((size_t)2 * kDim * kDim * 2);  // Wkv^T
  bf16_t* wpT = (bf16_t*)alloc((size_t)kDim * kDim * 2);       // Wproj^T
  bf16_t* qb = (bf16_t*)alloc((size_t)kM * kDim * 2);          // q [bh][N][d]
  bf16_t* kT = (bf16_t*)alloc((size_t)kM * kDim * 2);          // k [bh][d][N]
  bf16_t* vT = (bf16_t*)alloc((size_t)kM * kDim * 2);          // v [bh][d][N]
  float* kmean = (float*)alloc((size_t)kBH * kHd * 4);
  float* zb = (float*)alloc((size_t)kBH * kN * 4);
  bf16_t* kvT = (bf16_t*)alloc((size_t)kBH * kHd * kHd * 2);
  bf16_t* cb = (bf16_t*)alloc((size_t)kBH * kHd * kN * 2);     // conv planes
  bf16_t* yb = (bf16_t*)alloc((size_t)kM * kDim * 2);          // att+conv

  cvt_kernel<<<(kM * kDim) / 256, 256, 0, stream>>>(x, xb, kM * kDim);
  transpose_cvt_kernel<<<(kDim * kDim) / 256, 256, 0, stream>>>(Wq, wqT, kDim,
                                                                kDim);
  transpose_cvt_kernel<<<(2 * kDim * kDim) / 256, 256, 0, stream>>>(
      Wkv, wkvT, kDim, 2 * kDim);
  transpose_cvt_kernel<<<(kDim * kDim) / 256, 256, 0, stream>>>(Wproj, wpT,
                                                                kDim, kDim);
  gemm_qkv_kernel<<<(256 * 48) / 4, 128, 0, stream>>>(xb, wqT, wkvT, qb, kT,
                                                      vT);
  kmean_kernel<<<kBH, 256, 0, stream>>>(kT, kmean);
  z_kernel<<<kBH * 16, 256, 0, stream>>>(qb, kmean, zb);
  kv_kernel<<<kBH, 128, 0, stream>>>(kT, vT, kvT);
  dwconv_kernel<<<(kBH * kHd * kN) / 256, 256, 0, stream>>>(vT, dwc_w, dwc_b,
                                                            cb);
  att_kernel<<<dim3(kBH, 16), 128, 0, stream>>>(qb, kvT, zb, cb, yb);
  gemm_out_kernel<<<(256 * 16) / 4, 128, 0, stream>>>(yb, wpT, bproj, out);
}
